// GRU_RNN_44332652429391
// MI455X (gfx1250) — compile-verified, matrix operand order checked by reading, not run on hardware
//
#include <hip/hip_runtime.h>
#include <cstdint>
#include <cstddef>

// ---------------------------------------------------------------------------
// Types for WMMA (gfx1250 / CDNA5, wave32)
// ---------------------------------------------------------------------------
typedef _Float16 v16h __attribute__((ext_vector_type(16)));
typedef _Float16 h8   __attribute__((ext_vector_type(8)));
typedef float    v8f  __attribute__((ext_vector_type(8)));
typedef int      v4i  __attribute__((ext_vector_type(4)));

// pointer-to-vector types in explicit address spaces for the async builtin
typedef __attribute__((address_space(1))) v4i* gv4i_p;   // global
typedef __attribute__((address_space(3))) v4i* lv4i_p;   // LDS

#define V8F_ZERO {0.f,0.f,0.f,0.f,0.f,0.f,0.f,0.f}

__device__ __forceinline__ v8f wmma_f16(v16h a, v16h b, v8f c) {
  // D = A(16x32 f16) * B(32x16 f16) + C(16x16 f32)
  return __builtin_amdgcn_wmma_f32_16x16x32_f16(false, a, false, b, (short)0, c,
                                                false, false);
}

// --------------------------- async global->LDS ------------------------------
#ifndef HAVE_ASYNC_LDS
#if defined(__has_builtin)
#if __has_builtin(__builtin_amdgcn_global_load_async_to_lds_b128) && \
    __has_builtin(__builtin_amdgcn_s_wait_asynccnt)
#define HAVE_ASYNC_LDS 1
#else
#define HAVE_ASYNC_LDS 0
#endif
#else
#define HAVE_ASYNC_LDS 0
#endif
#endif

__device__ __forceinline__ void async_cp16(const void* g, void* l) {
#if HAVE_ASYNC_LDS
  __builtin_amdgcn_global_load_async_to_lds_b128((gv4i_p)g, (lv4i_p)l, 0, 0);
#else
  *(h8*)l = *(const h8*)g;          // synchronous fallback
#endif
}
__device__ __forceinline__ void async_wait0() {
#if HAVE_ASYNC_LDS
  __builtin_amdgcn_s_wait_asynccnt(0);
#endif
}

// ---------------------------------------------------------------------------
// Problem constants
// ---------------------------------------------------------------------------
#define T_STEPS 2000
#define BATCH   32
#define HID     1024
#define XDIM    352     // 351 conv features + 1 zero pad (K-tile alignment)
#define NKT     43      // K tiles of 32 (32 hidden + 11 x)
#define NWG     16      // persistent workgroups (== max cluster size)
#define XBYTES  (BATCH * XDIM * 2)   // 22,528 B per time step (f16)

// ---------------------------------------------------------------------------
// Workspace layout (bytes), all offsets 256B aligned. Total ~57.3 MB.
// ---------------------------------------------------------------------------
constexpr size_t OFF_WBIG = 0;                           // 4096*1376 f16
constexpr size_t OFF_XBUF = OFF_WBIG + 11272192ull;      // 2000*32*352 f16
constexpr size_t OFF_H32  = OFF_XBUF + 45056000ull;      // 2*32*1024 f32 (double buffer)
constexpr size_t OFF_H16  = OFF_H32  + 262144ull;        // 2*32*1024 f16 (double buffer)
constexpr size_t OFF_BETA = OFF_H16  + 131072ull;        // 4096 f32 fused biases
constexpr size_t OFF_G0   = OFF_BETA + 16384ull;         // 32*3072 f32 t==0 correction
constexpr size_t OFF_CMAT = OFF_G0   + 393216ull;        // 35*1024 f32 scale_out@out1
constexpr size_t OFF_DVEC = OFF_CMAT + 143360ull;        // 35 f32 fused out bias
constexpr size_t OFF_BAR  = OFF_DVEC + 256ull;           // 2 ints: arrive cnt, generation

// d_out layout: trj_out (32,2000,35) | y_last (32,1,35) | h_last (1,32,1024)
constexpr int OUT_O1 = 32 * 2000 * 35;   // 2,240,000
constexpr int OUT_O2 = OUT_O1 + 32 * 35; // 2,241,120

// ---------------------------------------------------------------------------
// Kernel 1: build W_big (4096 x 1376) in f16, pre-swizzled into WMMA B-fragment
// tile order. Column c = q*1024 + j, q = gate group {r, z, i_n, h_n}:
//   q=0 (r):  K[h] = Whh_r + Wcomb_r        K[x] = Wihx_r
//   q=1 (z):  K[h] = Whh_z + Wcomb_z        K[x] = Wihx_z
//   q=2 (in): K[h] = Wcomb_n                K[x] = Wihx_n
//   q=3 (hn): K[h] = Whh_n                  K[x] = 0
// Wcomb = Wihy @ out1_w (folded y-recurrence).
// Tile (nt,kt) = 512 f16; flat f = L*16 + e, lane L reads [L*16, L*16+16).
// ---------------------------------------------------------------------------
__global__ __launch_bounds__(256) void prep_weights(
    const float* __restrict__ Wih, const float* __restrict__ Whh,
    const float* __restrict__ out1w, _Float16* __restrict__ Wbig)
{
  const int nt = blockIdx.x;   // 0..255 (N tiles of 16)
  const int kt = blockIdx.y;   // 0..42  (K tiles of 32)
  const int f0 = threadIdx.x * 2;
  for (int f = f0; f < f0 + 2; ++f) {
    const int L = f >> 4, e = f & 15;
    const int v = e >> 1, p = e & 1, half = L >> 4, nl = L & 15;
    const int klocal = ((v & 4) ? 16 : 0) + half * 8 + (v & 3) * 2 + p;
    const int kg  = kt * 32 + klocal;
    const int col = nt * 16 + nl;
    const int q = col >> 10, j = col & 1023;
    float val = 0.f;
    if (kg < HID) {                                   // hidden part
      const int c = (q == 0) ? j : (q == 1) ? (1024 + j) : (2048 + j);
      if (q != 2) val = Whh[(size_t)c * HID + kg];    // Whh for r,z,hn
      if (q != 3) {                                   // + Wcomb for r,z,in
        const float* wr = Wih + (size_t)c * 386 + 351;
        float s = 0.f;
        #pragma unroll
        for (int m = 0; m < 35; ++m) s += wr[m] * out1w[m * HID + kg];
        val += s;
      }
    } else {                                          // x part
      const int cc = kg - HID;
      if (cc < 351 && q < 3) {
        const int c = q * 1024 + j;
        val = Wih[(size_t)c * 386 + cc];
      }
    }
    Wbig[((size_t)nt * NKT + kt) * 512 + f] = (_Float16)val;
  }
}

// ---------------------------------------------------------------------------
// Kernel 2: fused biases beta[4096], t==0 y correction G0[32][3072],
// Cmat = scale_out_w @ out1_w, dvec, zero h buffers + barrier counters
// (re-zeroed every launch -> graph-replay safe).
// ---------------------------------------------------------------------------
__global__ __launch_bounds__(256) void prep_misc(
    const float* __restrict__ Wih, const float* __restrict__ bih,
    const float* __restrict__ bhh, const float* __restrict__ out1w,
    const float* __restrict__ out1b, const float* __restrict__ y_in,
    const float* __restrict__ sow, const float* __restrict__ sob,
    char* __restrict__ ws)
{
  float*    beta = (float*)(ws + OFF_BETA);
  float*    G0   = (float*)(ws + OFF_G0);
  float*    Cmat = (float*)(ws + OFF_CMAT);
  float*    dvec = (float*)(ws + OFF_DVEC);
  float*    H32  = (float*)(ws + OFF_H32);
  _Float16* H16  = (_Float16*)(ws + OFF_H16);
  int*      bar  = (int*)(ws + OFF_BAR);

  int i = blockIdx.x * 256 + threadIdx.x;
  const int NB = 4096, NG = 32 * 3072, NC = 35 * HID, NH = 2 * BATCH * HID;

  if (i < NB) {                         // fused biases
    const int q = i >> 10, j = i & 1023;
    const int c = (q == 0) ? j : (q == 1) ? (1024 + j) : (2048 + j);
    float val;
    if (q == 3) val = bhh[2048 + j];
    else {
      const float* wr = Wih + (size_t)c * 386 + 351;
      float s = 0.f;
      #pragma unroll
      for (int m = 0; m < 35; ++m) s += wr[m] * out1b[m];
      val = bih[c] + s + ((q < 2) ? bhh[c] : 0.f);
    }
    beta[i] = val; return;
  }
  i -= NB;
  if (i < NG) {                         // t==0 correction: Wihy @ (y_in - out1_b)
    const int b = i / 3072, c = i % 3072;
    const float* wr = Wih + (size_t)c * 386 + 351;
    const float* yr = y_in + b * 35;
    float s = 0.f;
    #pragma unroll
    for (int m = 0; m < 35; ++m) s += wr[m] * (yr[m] - out1b[m]);
    G0[i] = s; return;
  }
  i -= NG;
  if (i < NC) {                         // Cmat[o][k]
    const int o = i >> 10, k = i & 1023;
    float s = 0.f;
    #pragma unroll
    for (int m = 0; m < 35; ++m) s += sow[o * 35 + m] * out1w[m * HID + k];
    Cmat[i] = s; return;
  }
  i -= NC;
  if (i < 35) {                         // dvec[o]
    float s = sob[i];
    #pragma unroll
    for (int m = 0; m < 35; ++m) s += sow[i * 35 + m] * out1b[m];
    dvec[i] = s; return;
  }
  i -= 35;
  if (i < NH) { H32[i] = 0.f; return; }
  i -= NH;
  if (i < NH) { H16[i] = (_Float16)0.f; return; }
  i -= NH;
  if (i < 2) { bar[i] = 0; return; }
}

// ---------------------------------------------------------------------------
// Kernel 3: conv front-end: scale_in affine, conv0 (39->117, k3 p4),
// conv1 (117->351, k3 dil3) through LDS; emits f16 rows Xbuf[t][b][0..351].
// ---------------------------------------------------------------------------
__global__ __launch_bounds__(256) void frontend(
    const float* __restrict__ x, const float* __restrict__ siw,
    const float* __restrict__ sib, const float* __restrict__ c0w,
    const float* __restrict__ c0b, const float* __restrict__ c1w,
    const float* __restrict__ c1b, _Float16* __restrict__ Xbuf)
{
  const int b  = blockIdx.x;
  const int t0 = blockIdx.y * 64;
  const int tid = threadIdx.x;
  __shared__ float sx[72 * 39];   // x_in,  times [t0-4 .. t0+67]
  __shared__ float sh[70 * 117];  // h0c,   times [t0   .. t0+69]

  for (int idx = tid; idx < 72 * 39; idx += 256) {    // x_in = scale_in(x), 0-pad
    const int i = idx / 39, c = idx % 39;
    const int tg = t0 - 4 + i;
    float v = 0.f;
    if (tg >= 0 && tg < T_STEPS) {
      const float* xr = x + ((size_t)b * T_STEPS + tg) * 39;
      const float* wr = siw + c * 39;
      float s = sib[c];
      #pragma unroll
      for (int cc = 0; cc < 39; ++cc) s += wr[cc] * xr[cc];
      v = s;
    }
    sx[i * 39 + c] = v;
  }
  __syncthreads();
  for (int idx = tid; idx < 70 * 117; idx += 256) {   // conv0 (pad 4 -> src = i+j)
    const int i = idx / 117, m = idx % 117;
    float s = c0b[m];
    for (int jj = 0; jj < 3; ++jj) {
      const float* xr = &sx[(i + jj) * 39];
      const float* wr = c0w + (size_t)m * 39 * 3 + jj;
      #pragma unroll
      for (int cc = 0; cc < 39; ++cc) s += wr[cc * 3] * xr[cc];
    }
    sh[i * 117 + m] = s;
  }
  __syncthreads();
  for (int idx = tid; idx < 64 * XDIM; idx += 256) {  // conv1 (dil 3) + f16 store
    const int i = idx / XDIM, o = idx % XDIM;
    const int t = t0 + i;
    if (t >= T_STEPS) continue;
    float v = 0.f;
    if (o < 351) {
      float s = c1b[o];
      for (int kk = 0; kk < 3; ++kk) {
        const float* hr = &sh[(i + 3 * kk) * 117];
        const float* wr = c1w + (size_t)o * 117 * 3 + kk;
        for (int m = 0; m < 117; ++m) s += wr[m * 3] * hr[m];
      }
      v = s;
    }
    Xbuf[((size_t)t * BATCH + b) * XDIM + o] = (_Float16)v;
  }
}

// ---------------------------------------------------------------------------
// Kernel 4: persistent GRU. 16 WGs x 8 wave32. WG g owns hidden [g*64,g*64+64).
// Per step: WMMA GEMM [h;x](32x1376) x W_big^T -> r/z/i_n/h_n, activations,
// double-buffered h, cluster barrier + atomic grid barrier, fused outputs.
// x features for step t+1 are staged into LDS with async global->LDS copies
// overlapped with activation/barrier/next h-GEMM.
// ---------------------------------------------------------------------------
union AFrag { v16h v; h8 h[2]; };

__device__ __forceinline__ void grid_barrier(int* cnt, int* gen, int target) {
  __syncthreads();
  if (threadIdx.x == 0) {
    __threadfence();                           // release slice writes
    __builtin_amdgcn_s_cluster_barrier();      // s_barrier_signal/wait -3 (NOP if uncluster)
    if (atomicAdd(cnt, 1) == NWG - 1) {
      atomicExch(cnt, 0);
      __threadfence();
      atomicAdd(gen, 1);
    } else {
      while (*((volatile int*)gen) < target) __builtin_amdgcn_s_sleep(1);
    }
  }
  __syncthreads();
  __threadfence();                             // acquire other slices
}

__global__ __launch_bounds__(256) void gru_persistent(
    char* __restrict__ ws, const float* __restrict__ out1w,
    const float* __restrict__ out1b, float* __restrict__ dout)
{
  const _Float16* Wbig = (const _Float16*)(ws + OFF_WBIG);
  const _Float16* Xbuf = (const _Float16*)(ws + OFF_XBUF);
  float*    H32  = (float*)(ws + OFF_H32);
  _Float16* H16  = (_Float16*)(ws + OFF_H16);
  const float* beta = (const float*)(ws + OFF_BETA);
  const float* G0   = (const float*)(ws + OFF_G0);
  const float* Cmat = (const float*)(ws + OFF_CMAT);
  const float* dvec = (const float*)(ws + OFF_DVEC);
  int* bar = (int*)(ws + OFF_BAR);

  __shared__ __align__(16) _Float16 sxin[2][BATCH * XDIM];  // x tiles, double buffered

  const int g   = blockIdx.x;          // 0..15
  const int tid = threadIdx.x;
  const int w   = tid >> 5, l = tid & 31;
  const int mt  = w & 1;               // batch tile (rows mt*16..+16)
  const int jt  = w >> 1;              // hidden sub-tile 0..3
  const int nl  = l & 15, half = l >> 4;
  const int jbase = g * 64 + jt * 16;
  const int ntb   = g * 4 + jt;        // n-tile within a gate group
  const int arow  = mt * 16 + nl;      // A-fragment source row (batch)
  const int j     = jbase + nl;        // this lane's hidden column

  // loop-invariant hoists: per-lane B-tile base pointers and fused biases
  const _Float16* const bb0 = Wbig + ((size_t)(0 * 64 + ntb) * NKT) * 512 + (size_t)l * 16;
  const _Float16* const bb1 = Wbig + ((size_t)(1 * 64 + ntb) * NKT) * 512 + (size_t)l * 16;
  const _Float16* const bb2 = Wbig + ((size_t)(2 * 64 + ntb) * NKT) * 512 + (size_t)l * 16;
  const _Float16* const bb3 = Wbig + ((size_t)(3 * 64 + ntb) * NKT) * 512 + (size_t)l * 16;
  const float cb0 = beta[j], cb1 = beta[1024 + j];
  const float cb2 = beta[2048 + j], cb3 = beta[3072 + j];

  // prime LDS with the t=0 x tile (contiguous 22.5 KB)
  {
    const char* gsrc = (const char*)Xbuf;
    char* ldst = (char*)&sxin[0][0];
    for (int off = tid * 16; off < XBYTES; off += 256 * 16)
      async_cp16(gsrc + off, ldst + off);
  }

  int par = 0;                         // h_{t-1} lives in buffer `par`
  for (int t = 0; t < T_STEPS; ++t) {
    v8f acc0 = V8F_ZERO, acc1 = V8F_ZERO, acc2 = V8F_ZERO, acc3 = V8F_ZERO;

    // ---- hidden part: kt = 0..31, 4 gate groups -------------------------
    {
      const _Float16* hp = H16 + (size_t)par * BATCH * HID + (size_t)arow * HID + half * 8;
      const _Float16* p0 = bb0;
      const _Float16* p1 = bb1;
      const _Float16* p2 = bb2;
      const _Float16* p3 = bb3;
      #pragma unroll 2
      for (int kt = 0; kt < 32; ++kt) {
        AFrag a, fb;
        a.h[0] = *(const h8*)(hp);
        a.h[1] = *(const h8*)(hp + 16);
        hp += 32;
        fb.h[0] = *(const h8*)(p0); fb.h[1] = *(const h8*)(p0 + 8); p0 += 512;
        acc0 = wmma_f16(a.v, fb.v, acc0);
        fb.h[0] = *(const h8*)(p1); fb.h[1] = *(const h8*)(p1 + 8); p1 += 512;
        acc1 = wmma_f16(a.v, fb.v, acc1);
        fb.h[0] = *(const h8*)(p2); fb.h[1] = *(const h8*)(p2 + 8); p2 += 512;
        acc2 = wmma_f16(a.v, fb.v, acc2);
        fb.h[0] = *(const h8*)(p3); fb.h[1] = *(const h8*)(p3 + 8); p3 += 512;
        acc3 = wmma_f16(a.v, fb.v, acc3);
      }
    }

    // ---- x part: kt = 32..42, 3 gate groups (h_n has no x term) ---------
    async_wait0();        // our async fills of sxin[t&1] (issued last step) done
    __syncthreads();      // ... and everyone else's
    {
      const _Float16* xp = &sxin[t & 1][arow * XDIM + half * 8];
      const _Float16* p0 = bb0 + (size_t)32 * 512;
      const _Float16* p1 = bb1 + (size_t)32 * 512;
      const _Float16* p2 = bb2 + (size_t)32 * 512;
      #pragma unroll 1
      for (int kt = 0; kt < 11; ++kt) {
        AFrag a, fb;
        a.h[0] = *(const h8*)(xp);            // ds_load_b128 from LDS
        a.h[1] = *(const h8*)(xp + 16);
        xp += 32;
        fb.h[0] = *(const h8*)(p0); fb.h[1] = *(const h8*)(p0 + 8); p0 += 512;
        acc0 = wmma_f16(a.v, fb.v, acc0);
        fb.h[0] = *(const h8*)(p1); fb.h[1] = *(const h8*)(p1 + 8); p1 += 512;
        acc1 = wmma_f16(a.v, fb.v, acc1);
        fb.h[0] = *(const h8*)(p2); fb.h[1] = *(const h8*)(p2 + 8); p2 += 512;
        acc2 = wmma_f16(a.v, fb.v, acc2);
      }
    }

    // ---- kick off async stage of next step's x tile ---------------------
    if (t + 1 < T_STEPS) {
      const char* gsrc = (const char*)(Xbuf + (size_t)(t + 1) * BATCH * XDIM);
      char* ldst = (char*)&sxin[(t + 1) & 1][0];
      for (int off = tid * 16; off < XBYTES; off += 256 * 16)
        async_cp16(gsrc + off, ldst + off);
    }

    // ---- activations + h update (each (m,j) owned by this wave) ---------
    const float*    H32r = H32 + (size_t)par * BATCH * HID;
    float*          H32w = H32 + (size_t)(par ^ 1) * BATCH * HID;
    _Float16*       H16w = H16 + (size_t)(par ^ 1) * BATCH * HID;
    #pragma unroll
    for (int r = 0; r < 8; ++r) {
      const int m = mt * 16 + half * 8 + r;      // C-fragment row mapping
      const float hprev = H32r[m * HID + j];
      float cr = 0.f, cz = 0.f, cn = 0.f;
      if (t == 0) {                              // y_in vs folded-y correction
        const float* Gp = G0 + (size_t)m * 3072;
        cr = Gp[j]; cz = Gp[1024 + j]; cn = Gp[2048 + j];
      }
      const float ar  = acc0[r] + cb0 + cr;
      const float az  = acc1[r] + cb1 + cz;
      const float ain = acc2[r] + cb2 + cn;
      const float ahn = acc3[r] + cb3;
      const float rg = 1.f / (1.f + __expf(-ar));
      const float zg = 1.f / (1.f + __expf(-az));
      const float pre = ain + rg * ahn;
      const float nn  = 1.f - 2.f / (1.f + __expf(2.f * pre));   // tanh
      const float hnew = (1.f - zg) * nn + zg * hprev;
      H32w[m * HID + j] = hnew;
      H16w[m * HID + j] = (_Float16)hnew;
    }

    grid_barrier(bar, bar + 1, t + 1);

    // ---- fused output: trj_out[b,t,:] = h_t @ Cmat^T + dvec (2 rows/WG) --
    const float* Hnew = H32 + (size_t)(par ^ 1) * BATCH * HID;
    for (int d = w; d < 70; d += 8) {
      const int bl = d / 35, o = d % 35;
      const int b  = g * 2 + bl;
      const float* hb = Hnew + (size_t)b * HID;
      const float* cm = Cmat + (size_t)o * HID;
      float s = 0.f;
      for (int k = l; k < HID; k += 32) s += hb[k] * cm[k];
      #pragma unroll
      for (int off = 16; off; off >>= 1) s += __shfl_down(s, off, 32);
      if (l == 0) dout[((size_t)b * T_STEPS + t) * 35 + o] = s + dvec[o];
      if (t == T_STEPS - 1) {                    // y_last uses raw out1 head
        const float* ow = out1w + (size_t)o * HID;
        float s2 = 0.f;
        for (int k = l; k < HID; k += 32) s2 += hb[k] * ow[k];
        #pragma unroll
        for (int off = 16; off; off >>= 1) s2 += __shfl_down(s2, off, 32);
        if (l == 0) dout[OUT_O1 + b * 35 + o] = s2 + out1b[o];
      }
    }
    par ^= 1;
  }

  // h_last: WG g owns hidden slice [g*64, g*64+64) for all batch rows
  const float* Hfin = H32 + (size_t)par * BATCH * HID;
  for (int idx = tid; idx < BATCH * 64; idx += 256) {
    const int m = idx >> 6, jl = idx & 63;
    dout[OUT_O2 + m * HID + g * 64 + jl] = Hfin[m * HID + g * 64 + jl];
  }
}

// ---------------------------------------------------------------------------
// Host launcher (graph-capture safe: stream-only, deterministic)
// ---------------------------------------------------------------------------
extern "C" void kernel_launch(void* const* d_in, const int* in_sizes, int n_in,
                              void* d_out, int out_size, void* d_ws, size_t ws_size,
                              hipStream_t stream) {
  const float* x    = (const float*)d_in[0];
  const float* y_in = (const float*)d_in[1];
  const float* siw  = (const float*)d_in[2];
  const float* sib  = (const float*)d_in[3];
  const float* c0w  = (const float*)d_in[4];
  const float* c0b  = (const float*)d_in[5];
  const float* c1w  = (const float*)d_in[6];
  const float* c1b  = (const float*)d_in[7];
  const float* Wih  = (const float*)d_in[8];
  const float* Whh  = (const float*)d_in[9];
  const float* bih  = (const float*)d_in[10];
  const float* bhh  = (const float*)d_in[11];
  const float* o1w  = (const float*)d_in[12];
  const float* o1b  = (const float*)d_in[13];
  const float* sow  = (const float*)d_in[14];
  const float* sob  = (const float*)d_in[15];
  char*  ws  = (char*)d_ws;
  float* out = (float*)d_out;

  prep_weights<<<dim3(256, NKT), 256, 0, stream>>>(
      Wih, Whh, o1w, (_Float16*)(ws + OFF_WBIG));
  prep_misc<<<1053, 256, 0, stream>>>(
      Wih, bih, bhh, o1w, o1b, y_in, sow, sob, ws);
  frontend<<<dim3(32, 32), 256, 0, stream>>>(
      x, siw, sib, c0w, c0b, c1w, c1b, (_Float16*)(ws + OFF_XBUF));
  gru_persistent<<<NWG, 256, 0, stream>>>(ws, o1w, o1b, out);
}
